// QuantumKernelQuanvolution_65481071403004
// MI455X (gfx1250) — matrix operand order repaired and checked by reading, not compiled, not run on hardware
//
#include <hip/hip_runtime.h>
#include <cstdint>
#include <cstddef>

typedef _Float16 v16h __attribute__((ext_vector_type(16)));
typedef _Float16 v8h  __attribute__((ext_vector_type(8)));
typedef float    v8f  __attribute__((ext_vector_type(8)));

#define IMG_ELEMS 784   // 28*28 input pixels per image
#define KDIM      784   // 4*14*14 features per image
#define KP        800   // K padded to multiple of 32 (25 WMMA k-steps)

// ---------------------------------------------------------------------------
// Phase 1: quanvolution features (conv 1->4, k=2, s=2) + exact f32 row norms.
// One 128-thread block per image. Stores f16 features, zero-padded to KP.
// ---------------------------------------------------------------------------
__global__ __launch_bounds__(128)
void quanv_features(const float* __restrict__ src,   // [N,1,28,28]
                    const float* __restrict__ cw,    // [4,1,2,2]
                    const float* __restrict__ cb,    // [4]
                    _Float16* __restrict__ feat,     // [N, KP] f16
                    float* __restrict__ norms)       // [N] f32 sum(f^2)
{
    const int n = blockIdx.x;
    const float* img = src + (size_t)n * IMG_ELEMS;
    _Float16* dst = feat + (size_t)n * KP;

    float w[16], b[4];
#pragma unroll
    for (int i = 0; i < 16; ++i) w[i] = cw[i];
#pragma unroll
    for (int i = 0; i < 4;  ++i) b[i] = cb[i];

    float acc = 0.0f;
    for (int d = threadIdx.x; d < KP; d += 128) {
        float v = 0.0f;
        if (d < KDIM) {
            const int c = d / 196;
            const int r = d % 196;
            const int i = r / 14;
            const int j = r % 14;
            const float* p = img + (2 * i) * 28 + 2 * j;
            v = w[c * 4 + 0] * p[0]  + w[c * 4 + 1] * p[1]
              + w[c * 4 + 2] * p[28] + w[c * 4 + 3] * p[29] + b[c];
            acc += v * v;
        }
        dst[d] = (_Float16)v;   // padding lanes write 0
    }

    __shared__ float red[128];
    red[threadIdx.x] = acc;
    __syncthreads();
#pragma unroll
    for (int s = 64; s > 0; s >>= 1) {
        if ((int)threadIdx.x < s) red[threadIdx.x] += red[threadIdx.x + s];
        __syncthreads();
    }
    if (threadIdx.x == 0) norms[n] = red[0];
}

// ---------------------------------------------------------------------------
// Phase 2: Gram matrix via v_wmma_f32_16x16x32_f16 + RBF epilogue.
// 256 threads = 8 waves; waves tiled 2(M) x 4(N); each wave owns a 64x64
// output macro-tile = 4x4 WMMA fragments. Block tile: 128(M) x 256(N).
// A/B stream from L2 (both feature matrices fit in 192MB L2 with huge room).
// ---------------------------------------------------------------------------
__global__ __launch_bounds__(256)
void gram_wmma(const _Float16* __restrict__ fxh,  // [N, KP]
               const _Float16* __restrict__ fyh,  // [M, KP]
               const float* __restrict__ x2,      // [N]
               const float* __restrict__ y2,      // [M]
               float* __restrict__ out,           // [N, M]
               int Mcols)
{
    const int lane = threadIdx.x & 31;
    const int wave = threadIdx.x >> 5;
    const int g    = lane >> 4;   // half-wave group (0/1)
    const int hl   = lane & 15;

    const int rowBlk = blockIdx.x * 128 + (wave & 1) * 64;  // x-row base (M dim of out)
    const int colBlk = blockIdx.y * 256 + (wave >> 1) * 64; // y-row base (N dim of out)

    v8f acc[4][4] = {};

    // Per-lane row bases (A fragment: lane holds row rowBlk+mt*16+hl;
    //                     B fragment: lane holds fy row colBlk+nt*16+hl).
    const _Float16* aBase = fxh + (size_t)(rowBlk + hl) * KP;
    const _Float16* bBase = fyh + (size_t)(colBlk + hl) * KP;

#pragma unroll 1
    for (int k = 0; k < KP; k += 32) {
        v16h afr[4], bfr[4];

        // A 16x32 f16 layout: VGPR0-3 hold K = g*8 + 0..7, VGPR4-7 hold K = 16 + g*8 + 0..7
#pragma unroll
        for (int mt = 0; mt < 4; ++mt) {
            const _Float16* p = aBase + (size_t)mt * 16 * KP + k + g * 8;
            union { v16h v; v8h h[2]; } u;
            u.h[0] = *(const v8h*)(p);
            u.h[1] = *(const v8h*)(p + 16);
            afr[mt] = u.v;
        }
        // B 32x16 f16 layout: VGPR0-3 hold K = g*8 + 0..7, VGPR4-7 hold K = 16 + g*8 + 0..7
#pragma unroll
        for (int nt = 0; nt < 4; ++nt) {
            const _Float16* p = bBase + (size_t)nt * 16 * KP + k + g * 8;
            union { v16h v; v8h h[2]; } u;
            u.h[0] = *(const v8h*)(p);
            u.h[1] = *(const v8h*)(p + 16);
            bfr[nt] = u.v;
        }

#pragma unroll
        for (int mt = 0; mt < 4; ++mt)
#pragma unroll
            for (int nt = 0; nt < 4; ++nt)
                acc[mt][nt] = __builtin_amdgcn_wmma_f32_16x16x32_f16(
                    false, afr[mt], false, bfr[nt],
                    (short)0, acc[mt][nt], false, false);
    }

    // Epilogue: out = exp(-max(x2 + y2 - 2*xy, 0)) (GAMMA = 1)
    // C/D layout: VGPR v -> M = v + g*8 (mod tile), lane hl -> N.
#pragma unroll
    for (int mt = 0; mt < 4; ++mt) {
#pragma unroll
        for (int nt = 0; nt < 4; ++nt) {
            const int col = colBlk + nt * 16 + hl;
            const float yn = y2[col];
#pragma unroll
            for (int v = 0; v < 8; ++v) {
                const int row = rowBlk + mt * 16 + g * 8 + v;
                const float xy = acc[mt][nt][v];
                float sq = x2[row] + yn - 2.0f * xy;
                sq = fmaxf(sq, 0.0f);
                out[(size_t)row * Mcols + col] = __expf(-sq);
            }
        }
    }
}

// ---------------------------------------------------------------------------
// Launch
// ---------------------------------------------------------------------------
extern "C" void kernel_launch(void* const* d_in, const int* in_sizes, int n_in,
                              void* d_out, int out_size, void* d_ws, size_t ws_size,
                              hipStream_t stream) {
    const float* x  = (const float*)d_in[0];
    const float* y  = (const float*)d_in[1];
    const float* cw = (const float*)d_in[2];
    const float* cb = (const float*)d_in[3];
    float* out = (float*)d_out;

    const int N = in_sizes[0] / IMG_ELEMS;  // 4096
    const int M = in_sizes[1] / IMG_ELEMS;  // 4096

    uint8_t* ws = (uint8_t*)d_ws;
    const size_t fxBytes = (size_t)N * KP * sizeof(_Float16);
    const size_t fyBytes = (size_t)M * KP * sizeof(_Float16);
    _Float16* fxh = (_Float16*)(ws);
    _Float16* fyh = (_Float16*)(ws + fxBytes);
    float*    x2  = (float*)(ws + fxBytes + fyBytes);
    float*    y2  = (float*)(ws + fxBytes + fyBytes + (size_t)N * sizeof(float));

    quanv_features<<<dim3(N), 128, 0, stream>>>(x, cw, cb, fxh, x2);
    quanv_features<<<dim3(M), 128, 0, stream>>>(y, cw, cb, fyh, y2);

    dim3 grid(N / 128, M / 256);   // 32 x 16 blocks for 4096x4096
    gram_wmma<<<grid, 256, 0, stream>>>(fxh, fyh, x2, y2, out, M);
}
